// SelfAttention_25323127177527
// MI455X (gfx1250) — matrix operand order repaired and checked by reading, not compiled, not run on hardware
//
#include <hip/hip_runtime.h>
#include <stdint.h>

typedef __attribute__((ext_vector_type(16))) _Float16 v16h;
typedef __attribute__((ext_vector_type(8)))  _Float16 v8h;
typedef __attribute__((ext_vector_type(8)))  float    v8f;

#define NB   4
#define NH   16
#define SEQ  2048
#define DH   128
#define Bc   64            // keys per chunk
#define KT_STR 136         // halves per K row in f16 LDS (128 + 8 pad)
#define VT_STR 72          // halves per V^T row in f16 LDS (64 keys + 8 pad)
#define PT_STR 72          // halves per P row in LDS (64 + 8 pad)
#define NEGBIG (-1000000000.0f)

__global__ __launch_bounds__(256, 1)
void fattn_kernel(const float* __restrict__ Q, const float* __restrict__ K,
                  const float* __restrict__ V, const int* __restrict__ mask,
                  float* __restrict__ out)
{
    // single f32 staging buffers, filled by async global->LDS DMA
    __shared__ __align__(16) float    kf[Bc * DH];            // K chunk f32 [key][d]
    __shared__ __align__(16) float    vf[Bc * DH];            // V chunk f32 [key][d]
    // f16 compute tiles (each element converted exactly once per WG)
    __shared__ __align__(16) _Float16 kt[Bc * KT_STR];        // K f16 [key][d]
    __shared__ __align__(16) _Float16 vt[DH * VT_STR];        // V f16 transposed [d][key]
    __shared__ __align__(16) _Float16 pt[8 * 16 * PT_STR];    // per-wave P staging

    const int tid  = threadIdx.x;
    const int wid  = tid >> 5;
    const int lane = tid & 31;
    const int g    = lane >> 4;   // half-wave
    const int n    = lane & 15;   // N col (B/C/D layouts) / M row (A layout)

    const int nqblk = SEQ / 128;
    const int bh    = blockIdx.x / nqblk;
    const int qblk  = blockIdx.x % nqblk;
    const int b     = bh / NH;
    const int qbase = qblk * 128 + wid * 16;

    const size_t head_off = (size_t)bh * SEQ * DH;
    const float* Qh = Q + head_off;
    const float* Kh = K + head_off;
    const float* Vh = V + head_off;
    float*       Oh = out + head_off;
    const int*   Mb = mask + (size_t)b * SEQ * SEQ;

    // fire-and-forget async stage of one 64-key chunk (row-major f32, b128)
    auto stage = [&](int kc) {
        #pragma unroll
        for (int j = 0; j < 8; ++j) {
            const int key  = wid * 8 + j;
            const int voff = (((kc + key) * DH) + lane * 4) * 4;
            uint32_t ldsk = (uint32_t)(uintptr_t)&kf[key * DH + lane * 4];
            asm volatile("global_load_async_to_lds_b128 %0, %1, %2"
                         :: "v"(ldsk), "v"(voff), "s"(Kh) : "memory");
            uint32_t ldsv = (uint32_t)(uintptr_t)&vf[key * DH + lane * 4];
            asm volatile("global_load_async_to_lds_b128 %0, %1, %2"
                         :: "v"(ldsv), "v"(voff), "s"(Vh) : "memory");
        }
    };

    // ---- Q tile (16 x 128) as 4 A-fragments (f16), resident in VGPRs.
    // A layout: lane m = lane%16; half h: K = 16*g + h
    v16h qfrag[4];
    {
        const float* qp = Qh + (size_t)(qbase + n) * DH;
        #pragma unroll
        for (int dk = 0; dk < 4; ++dk) {
            const float* p0 = qp + dk * 32 + 16 * g;
            const float* p1 = p0 + 8;
            #pragma unroll
            for (int i = 0; i < 8; ++i) {
                qfrag[dk][i]     = (_Float16)p0[i];
                qfrag[dk][8 + i] = (_Float16)p1[i];
            }
        }
    }

    // all-ones B fragment: rowsum(P) = P x ones via WMMA (layout-independent)
    v16h ones;
    #pragma unroll
    for (int i = 0; i < 16; ++i) ones[i] = (_Float16)1.0f;

    // single mask base pointer; (r,t) offsets are unrolled-constant immediates
    const int* mptr = Mb + (size_t)(qbase + 8 * g) * SEQ + n;

    v8f zero = {};
    v8f o[8];
    #pragma unroll
    for (int s = 0; s < 8; ++s) o[s] = zero;
    float mrun[8], lrun[8];
    #pragma unroll
    for (int r = 0; r < 8; ++r) { mrun[r] = -INFINITY; lrun[r] = 0.0f; }

    // prologue: stage chunk 0
    stage(0);
    asm volatile("s_wait_asynccnt 0" ::: "memory");
    __syncthreads();

    for (int kc = 0; kc < SEQ; kc += Bc, mptr += Bc) {
        // ---- cooperative f32 -> f16 conversion pass (each element once per WG)
        #pragma unroll
        for (int i = 0; i < 8; ++i) {
            const int flat = i * 256 + tid;
            {   // K: contiguous read, contiguous b64 f16 store
                const int key = flat >> 5, c4 = flat & 31;
                const float4 kv = *(const float4*)&kf[key * DH + c4 * 4];
                _Float16* kd = &kt[key * KT_STR + c4 * 4];
                kd[0] = (_Float16)kv.x; kd[1] = (_Float16)kv.y;
                kd[2] = (_Float16)kv.z; kd[3] = (_Float16)kv.w;
            }
            {   // V: conflict-free strided reads, b64 store into transposed row
                const int d = flat & 127, k16 = flat >> 7;
                _Float16* vd = &vt[d * VT_STR + k16 * 4];
                #pragma unroll
                for (int j = 0; j < 4; ++j)
                    vd[j] = (_Float16)vf[(k16 * 4 + j) * DH + d];
            }
        }
        __syncthreads();   // f16 tiles ready; f32 buffers fully consumed

        // prefetch next chunk into the (now free) f32 buffers
        if (kc + Bc < SEQ) {
            stage(kc + Bc);
            if (tid < 128)
                __builtin_prefetch(Mb + (size_t)(qblk * 128 + tid) * SEQ + kc + Bc, 0, 1);
        }

        // ---- S = Q * K^T : 16 flattened steps (4 tiles x 4 K-steps),
        // explicitly double-buffered so loads for step u+1 precede WMMA u
        v8f acc[4];
        #pragma unroll
        for (int t = 0; t < 4; ++t) acc[t] = zero;
        {
            v8h bl[2], bh[2];
            const _Float16* p0 = &kt[n * KT_STR + 16 * g];
            bl[0] = *(const v8h*)(p0);
            bh[0] = *(const v8h*)(p0 + 8);
            #pragma unroll
            for (int u = 0; u < 16; ++u) {
                const int cur = u & 1;
                if (u < 15) {
                    const int t = (u + 1) >> 2, dk = (u + 1) & 3;
                    const _Float16* p = &kt[(t * 16 + n) * KT_STR + dk * 32 + 16 * g];
                    bl[cur ^ 1] = *(const v8h*)(p);
                    bh[cur ^ 1] = *(const v8h*)(p + 8);
                }
                v16h bfrag;
                #pragma unroll
                for (int i = 0; i < 8; ++i) { bfrag[i] = bl[cur][i]; bfrag[8 + i] = bh[cur][i]; }
                acc[u >> 2] = __builtin_amdgcn_wmma_f32_16x16x32_f16(
                    false, qfrag[u & 3], false, bfrag, (short)0, acc[u >> 2], false, false);
            }
        }

        // ---- mask + online softmax (row M = r + 8*g, col N = n)
        float pv[4][8], scale_r[8];
        #pragma unroll
        for (int r = 0; r < 8; ++r) {
            float sv[4];
            #pragma unroll
            for (int t = 0; t < 4; ++t)
                sv[t] = (mptr[r * SEQ + t * 16] != 0) ? acc[t][r] : NEGBIG;
            float rmax = fmaxf(fmaxf(sv[0], sv[1]), fmaxf(sv[2], sv[3]));
            #pragma unroll
            for (int d = 8; d >= 1; d >>= 1)
                rmax = fmaxf(rmax, __shfl_xor(rmax, d, 32));
            const float mnew = fmaxf(mrun[r], rmax);
            scale_r[r] = __expf(mrun[r] - mnew);
            mrun[r] = mnew;
            #pragma unroll
            for (int t = 0; t < 4; ++t)
                pv[t][r] = __expf(sv[t] - mnew);
            #pragma unroll
            for (int s = 0; s < 8; ++s) o[s][r] *= scale_r[r];
        }

        // ---- P (C-layout f32) -> LDS row-major f16 -> reload as A-fragments
        _Float16* pw = &pt[wid * 16 * PT_STR];
        #pragma unroll
        for (int r = 0; r < 8; ++r) {
            const int m = r + 8 * g;
            #pragma unroll
            for (int t = 0; t < 4; ++t)
                pw[m * PT_STR + t * 16 + n] = (_Float16)pv[t][r];
        }
        asm volatile("s_wait_dscnt 0" ::: "memory");  // wave-local LDS RAW

        v16h pa[2];   // A-fragments for key halves 0..31 and 32..63
        {
            const _Float16* pr = &pw[n * PT_STR];
            #pragma unroll
            for (int hseg = 0; hseg < 2; ++hseg) {
                const v8h alo = *(const v8h*)(pr + hseg * 32 + 16 * g);
                const v8h ahi = *(const v8h*)(pr + hseg * 32 + 16 * g + 8);
                #pragma unroll
                for (int i = 0; i < 8; ++i) { pa[hseg][i] = alo[i]; pa[hseg][8 + i] = ahi[i]; }
            }
        }

        // ---- row sums of P via ones-WMMA: lsum[r] = sum_k P[row r][k]
        // (uses the same f16-rounded P as the P*V numerator)
        v8f lsum = zero;
        lsum = __builtin_amdgcn_wmma_f32_16x16x32_f16(
            false, pa[0], false, ones, (short)0, lsum, false, false);
        lsum = __builtin_amdgcn_wmma_f32_16x16x32_f16(
            false, pa[1], false, ones, (short)0, lsum, false, false);
        #pragma unroll
        for (int r = 0; r < 8; ++r)
            lrun[r] = lrun[r] * scale_r[r] + lsum[r];

        // ---- O += P * V : 16 flattened steps (8 slices x 2 key-halves),
        // double-buffered like QK
        {
            v8h bl[2], bh[2];
            const _Float16* p0 = &vt[n * VT_STR + 16 * g];
            bl[0] = *(const v8h*)(p0);
            bh[0] = *(const v8h*)(p0 + 8);
            #pragma unroll
            for (int u = 0; u < 16; ++u) {
                const int cur = u & 1;
                if (u < 15) {
                    const int s = (u + 1) >> 1, hseg = (u + 1) & 1;
                    const _Float16* p = &vt[(s * 16 + n) * VT_STR + hseg * 32 + 16 * g];
                    bl[cur ^ 1] = *(const v8h*)(p);
                    bh[cur ^ 1] = *(const v8h*)(p + 8);
                }
                v16h bfrag;
                #pragma unroll
                for (int i = 0; i < 8; ++i) { bfrag[i] = bl[cur][i]; bfrag[8 + i] = bh[cur][i]; }
                o[u >> 1] = __builtin_amdgcn_wmma_f32_16x16x32_f16(
                    false, pa[u & 1], false, bfrag, (short)0, o[u >> 1], false, false);
            }
        }

        // async copies for chunk kc+Bc must land before next conversion pass
        asm volatile("s_wait_asynccnt 0" ::: "memory");
        __syncthreads();
    }

    // ---- epilogue: normalize by row sum, store fp32 (immediate offsets)
    {
        float* obase = Oh + (size_t)(qbase + 8 * g) * DH + n;
        #pragma unroll
        for (int r = 0; r < 8; ++r) {
            const float inv = 1.0f / lrun[r];
            #pragma unroll
            for (int s = 0; s < 8; ++s)
                obase[r * DH + s * 16] = o[s][r] * inv;
        }
    }
}

extern "C" void kernel_launch(void* const* d_in, const int* in_sizes, int n_in,
                              void* d_out, int out_size, void* d_ws, size_t ws_size,
                              hipStream_t stream) {
    const float* Q    = (const float*)d_in[0];
    const float* K    = (const float*)d_in[1];
    const float* V    = (const float*)d_in[2];
    const int*   mask = (const int*)d_in[3];
    float*       out  = (float*)d_out;
    (void)in_sizes; (void)n_in; (void)out_size; (void)d_ws; (void)ws_size;

    const int grid = NB * NH * (SEQ / 128);   // 1024 workgroups
    fattn_kernel<<<grid, 256, 0, stream>>>(Q, K, V, mask, out);
}
